// AllLoss_63093069578246
// MI455X (gfx1250) — matrix operand order repaired and checked by reading, not compiled, not run on hardware
//
#include <hip/hip_runtime.h>
#include <cstdint>

// ---------------------------------------------------------------------------
// Cosine-sim contrastive loss, fused flash-style on gfx1250 (MI455X).
//   loss = sum_i ( log(sum_j exp(s_ij)) - s_ii ),  s = (Q/|Q|)(R/|R|)^T
// Scores bounded in [-1,1] => exp() safe without max-shift, so partial
// exp-sums are additive (deterministic column-split, no atomics).
// GEMM in bf16 WMMA (v_wmma_f32_16x16x32_bf16); diagonal pos_i exact fp32.
// R tiles stream into LDS with GLOBAL_LOAD_ASYNC_TO_LDS_B128, double-buffered
// so the fill fully overlaps WMMA compute (ASYNCcnt + one barrier per chunk).
// ---------------------------------------------------------------------------

typedef __attribute__((ext_vector_type(16))) __bf16 v16bf;
typedef __attribute__((ext_vector_type(8)))  float  v8f;

union ABFrag {              // 16 bf16 (32B) viewed as two 16B loads
    v16bf v;
    uint4 q[2];
};

#define NROWS        8192
#define DDIM         256
#define MT           2       // M-tiles per wave (32 rows/wave)
#define BM           256     // rows per block (8 waves x 32)
#define BN_CHUNK     32      // R rows staged in LDS per chunk
#define COLS_PER_BLK 1024    // columns handled per block (32 chunks)
#define NCHUNK       (COLS_PER_BLK / BN_CHUNK)
#define LDS_STRIDE   264     // 256 + 8 pad halfwords -> 4-bank rotation/row
#define BUF_HW       (BN_CHUNK * LDS_STRIDE)   // halfwords per buffer (8448)

// float -> bf16, round-to-nearest-even (inputs are finite)
static __device__ __forceinline__ unsigned short f2bf(float x) {
    union { float f; uint32_t u; } v; v.f = x;
    uint32_t r = v.u + 0x7FFFu + ((v.u >> 16) & 1u);
    return (unsigned short)(r >> 16);
}

// ---------------------------------------------------------------------------
// Kernel 1: per-row norms, exact fp32 diagonal score, normalized bf16 rows.
// ---------------------------------------------------------------------------
__global__ __launch_bounds__(256)
void prep_kernel(const float* __restrict__ q, const float* __restrict__ r,
                 unsigned short* __restrict__ qh, unsigned short* __restrict__ rh,
                 float* __restrict__ pos) {
    const int i = blockIdx.x;
    const int t = threadIdx.x;
    const float qv = q[(size_t)i * DDIM + t];
    const float rv = r[(size_t)i * DDIM + t];

    __shared__ float s0[256], s1[256], s2[256];
    s0[t] = qv * qv; s1[t] = rv * rv; s2[t] = qv * rv;
    __syncthreads();
    for (int st = 128; st > 0; st >>= 1) {
        if (t < st) { s0[t] += s0[t + st]; s1[t] += s1[t + st]; s2[t] += s2[t + st]; }
        __syncthreads();
    }
    const float sq = s0[0], sr = s1[0], sqr = s2[0];
    const float qn = sqrtf(sq), rn = sqrtf(sr);
    const float invq = 1.0f / fmaxf(qn, 1e-20f);
    const float invr = 1.0f / fmaxf(rn, 1e-20f);
    qh[(size_t)i * DDIM + t] = f2bf(qv * invq);
    rh[(size_t)i * DDIM + t] = f2bf(rv * invr);
    if (t == 0) pos[i] = sqr / fmaxf(qn * rn, 1e-8f);   // torch eps semantics
}

// ---------------------------------------------------------------------------
// Kernel 2: fused GEMM + exp-sum.
// grid = (N/BM, N/COLS_PER_BLK); block = 256 threads (8 wave32).
// Wave w owns rows [blk.x*256 + 32w, +32) as 2 M-tiles held in 128 VGPRs.
// R chunks of 32 rows double-buffered in LDS via async-to-LDS DMA.
// partial[blk.y][row] = sum_{j in column range} exp(s_row,j)
// ---------------------------------------------------------------------------
__global__ __launch_bounds__(256)
void cosim_expsum_kernel(const unsigned short* __restrict__ qh,
                         const unsigned short* __restrict__ rh,
                         float* __restrict__ partial) {
    __shared__ unsigned short ldsB[2 * BUF_HW];   // 33,792 B total

    const int tid  = threadIdx.x;
    const int wave = tid >> 5;
    const int lane = tid & 31;
    const int half = lane >> 4;            // A/B lane-group select
    const int mn   = lane & 15;

    const int row0    = blockIdx.x * BM + wave * (16 * MT);
    const int colbase = blockIdx.y * COLS_PER_BLK;

    // ---- A tiles (2 x 16x256 bf16) into registers -------------------------
    // ISA 16-bit A 16x32 layout: lanes 0-15 -> k in {0..7,16..23},
    // lanes 16-31 -> k in {8..15,24..31}; two 16B contiguous chunks per lane.
    const int sel = half * 8;
    ABFrag a[MT][8];
    #pragma unroll
    for (int mt = 0; mt < MT; ++mt) {
        const uint4* qrow = (const uint4*)(qh + (size_t)(row0 + mt * 16 + mn) * DDIM);
        #pragma unroll
        for (int kt = 0; kt < 8; ++kt) {
            const int e0 = kt * 32 + sel;
            a[mt][kt].q[0] = qrow[(e0 >> 3) + 0];   // k = e0 .. e0+7
            a[mt][kt].q[1] = qrow[(e0 >> 3) + 2];   // k = e0+16 .. e0+23
        }
    }

    float accv[MT][8];
    #pragma unroll
    for (int mt = 0; mt < MT; ++mt)
        #pragma unroll
        for (int v = 0; v < 8; ++v) accv[mt][v] = 0.0f;

    // generic LDS pointer low 32 bits == LDS byte offset (flat LDS aperture)
    const uint32_t ldsbase = (uint32_t)(uintptr_t)&ldsB[0];

    // async fill of one 32-row chunk: 1024 x 16B transfers, 4 per thread
    auto issue_fill = [&](int buf, int cb) {
        #pragma unroll
        for (int it = 0; it < 4; ++it) {
            const int flat = tid + 256 * it;
            const int c = flat >> 5, off = flat & 31;
            const unsigned short* g = rh + (size_t)(cb + c) * DDIM + off * 8;
            const uint32_t l = ldsbase +
                (uint32_t)(buf * BUF_HW + c * LDS_STRIDE + off * 8) * 2u;
            asm volatile("global_load_async_to_lds_b128 %0, %1, off"
                         :: "v"(l), "v"((unsigned long long)(uintptr_t)g)
                         : "memory");
        }
    };

    // prologue: fill buffer 0 with chunk 0
    issue_fill(0, colbase);
    asm volatile("s_wait_asynccnt 0" ::: "memory");
    __syncthreads();

    const int sel2 = half * 8;             // B k-group per lane half
    for (int cc = 0; cc < NCHUNK; ++cc) {
        // kick off next chunk into the other buffer (overlaps compute)
        if (cc + 1 < NCHUNK)
            issue_fill((cc + 1) & 1, colbase + (cc + 1) * BN_CHUNK);

        const unsigned short* base = ldsB + (cc & 1) * BUF_HW;
        #pragma unroll
        for (int nt = 0; nt < BN_CHUNK / 16; ++nt) {
            v8f c0 = {0.f,0.f,0.f,0.f,0.f,0.f,0.f,0.f};
            v8f c1 = {0.f,0.f,0.f,0.f,0.f,0.f,0.f,0.f};
            const unsigned short* bbase = base + (nt * 16 + mn) * LDS_STRIDE;
            #pragma unroll
            for (int kt = 0; kt < 8; ++kt) {
                ABFrag b;                  // 32 contiguous bf16 (k-major)
                const int e0 = kt * 32 + sel2;
                b.q[0] = *(const uint4*)(bbase + e0);
                b.q[1] = *(const uint4*)(bbase + e0 + 16);
                // one B fragment feeds both M-tiles (2x LDS reuse)
                c0 = __builtin_amdgcn_wmma_f32_16x16x32_bf16(
                        false, a[0][kt].v, false, b.v, (short)0, c0, false, false);
                c1 = __builtin_amdgcn_wmma_f32_16x16x32_bf16(
                        false, a[1][kt].v, false, b.v, (short)0, c1, false, false);
            }
            #pragma unroll
            for (int v = 0; v < 8; ++v) {
                accv[0][v] += __expf(c0[v]);
                accv[1][v] += __expf(c1[v]);
            }
        }
        // my async fill done + everyone done reading / filling
        asm volatile("s_wait_asynccnt 0" ::: "memory");
        __syncthreads();
    }

    // reduce across the 16 lanes of each half (C layout: n = lane%16)
    #pragma unroll
    for (int off = 8; off > 0; off >>= 1)
        #pragma unroll
        for (int mt = 0; mt < MT; ++mt)
            #pragma unroll
            for (int v = 0; v < 8; ++v)
                accv[mt][v] += __shfl_xor(accv[mt][v], off, 32);

    // lane 0 holds rows +0..7 of each tile, lane 16 holds rows +8..15
    float* dst = partial + (size_t)blockIdx.y * NROWS + row0;
    if (lane == 0)
        #pragma unroll
        for (int mt = 0; mt < MT; ++mt)
            #pragma unroll
            for (int v = 0; v < 8; ++v) dst[mt * 16 + v] = accv[mt][v];
    if (lane == 16)
        #pragma unroll
        for (int mt = 0; mt < MT; ++mt)
            #pragma unroll
            for (int v = 0; v < 8; ++v) dst[mt * 16 + 8 + v] = accv[mt][v];
}

// ---------------------------------------------------------------------------
// Kernel 3: deterministic finalize: sum_i( log(sum_c partial[c][i]) - pos[i] )
// ---------------------------------------------------------------------------
__global__ __launch_bounds__(1024)
void finalize_kernel(const float* __restrict__ partial,
                     const float* __restrict__ pos,
                     float* __restrict__ out, int nchunks) {
    __shared__ float red[1024];
    const int t = threadIdx.x;
    float local = 0.0f;
    for (int i = t; i < NROWS; i += 1024) {
        float s = 0.0f;
        for (int c = 0; c < nchunks; ++c) s += partial[(size_t)c * NROWS + i];
        local += logf(s) - pos[i];
    }
    red[t] = local;
    __syncthreads();
    for (int st = 512; st > 0; st >>= 1) {
        if (t < st) red[t] += red[t + st];
        __syncthreads();
    }
    if (t == 0) out[0] = red[0];
}

// ---------------------------------------------------------------------------
extern "C" void kernel_launch(void* const* d_in, const int* in_sizes, int n_in,
                              void* d_out, int out_size, void* d_ws, size_t ws_size,
                              hipStream_t stream) {
    (void)n_in; (void)out_size; (void)ws_size; (void)in_sizes;
    const float* q = (const float*)d_in[0];
    const float* r = (const float*)d_in[1];
    float* out = (float*)d_out;

    // workspace layout (~8.5 MB):
    //   [0,4MB)  q-hat bf16   [4MB,8MB) r-hat bf16
    //   [8MB)    pos fp32[N]  then partial fp32[8][N]
    char* ws = (char*)d_ws;
    unsigned short* qh = (unsigned short*)ws;
    unsigned short* rh = (unsigned short*)(ws + (size_t)NROWS * DDIM * 2);
    float* pos     = (float*)(ws + (size_t)NROWS * DDIM * 4);
    float* partial = pos + NROWS;

    const int nchunks = NROWS / COLS_PER_BLK;   // 8 column splits

    prep_kernel<<<NROWS, 256, 0, stream>>>(q, r, qh, rh, pos);
    dim3 grid(NROWS / BM, nchunks);
    cosim_expsum_kernel<<<grid, 256, 0, stream>>>(qh, rh, partial);
    finalize_kernel<<<1, 1024, 0, stream>>>(partial, pos, out, nchunks);
}
